// GraphSAGE_56994216017995
// MI455X (gfx1250) — matrix operand order repaired and checked by reading, not compiled, not run on hardware
//
#include <hip/hip_runtime.h>
#include <hip/hip_bf16.h>
#include <stdint.h>
#include <stddef.h>

// ---------------------------------------------------------------------------
// Types for CDNA5 WMMA (wave32). 16x16x32 bf16 -> f32 accumulate.
// ---------------------------------------------------------------------------
typedef __bf16 bf16_t;
typedef __attribute__((ext_vector_type(16))) __bf16 v16bf;
typedef __attribute__((ext_vector_type(8)))  __bf16 v8bf;
typedef __attribute__((ext_vector_type(4)))  __bf16 v4bf;
typedef __attribute__((ext_vector_type(8)))  float  v8f;

#define HIDC 128   // hidden channels

// ---------------------------------------------------------------------------
// gfx1250 async global->LDS copy (ASYNCcnt-tracked, ISA ch.10 / 08_async_tensor).
// ldsOff is the workgroup-relative LDS byte offset (low 32 bits of the flat
// address of a __shared__ object), gaddr a per-lane 16B-aligned global address.
// ---------------------------------------------------------------------------
__device__ inline void async_copy_b128_to_lds(uint32_t ldsOff, const void* gaddr) {
    asm volatile("global_load_async_to_lds_b128 %0, %1, off"
                 :: "v"(ldsOff), "v"(gaddr)
                 : "memory");
}
__device__ inline void wait_asynccnt0() {
    asm volatile("s_wait_asynccnt 0" ::: "memory");
}

// ---------------------------------------------------------------------------
// A-fragment loader. 16x32 A fragment, contiguous k order used here (matches B):
//   lane 0-15  = row M=lane,     elems 0..15 -> K=kc+0..15
//   lane 16-31 = row M=lane-16,  elems 0..15 -> K=kc+16..31
// 'sub' = (lane>>4)*16 selects the half. Optionally converts f32 -> bf16 on load.
// ---------------------------------------------------------------------------
template <bool A_IS_F32>
__device__ inline v16bf load_a_frag(const void* __restrict__ Av, int row, int lda,
                                    int kc, int sub) {
    if constexpr (A_IS_F32) {
        const float* p = (const float*)Av + (size_t)row * lda + kc + sub;
        v8bf lo, hi;
#pragma unroll
        for (int i = 0; i < 8; ++i) lo[i] = (__bf16)p[i];
#pragma unroll
        for (int i = 0; i < 8; ++i) hi[i] = (__bf16)p[8 + i];
        return __builtin_shufflevector(lo, hi, 0, 1, 2, 3, 4, 5, 6, 7,
                                               8, 9, 10, 11, 12, 13, 14, 15);
    } else {
        const bf16_t* p = (const bf16_t*)Av + (size_t)row * lda + kc + sub;
        v8bf lo = *(const v8bf*)p;
        v8bf hi = *(const v8bf*)(p + 8);
        return __builtin_shufflevector(lo, hi, 0, 1, 2, 3, 4, 5, 6, 7,
                                               8, 9, 10, 11, 12, 13, 14, 15);
    }
}

// B-fragment loader (global path). B k order used here (KxN, K=32, matches A): lane 0-15 hold
// K=0..15 of column n=lane&15; lanes 16-31 hold K=16..31. With Bt stored
// transposed (Bt[n][k], row-major, row stride K), each lane reads 16
// contiguous bf16 (32 B).
__device__ inline v16bf load_b_frag(const bf16_t* __restrict__ Bt, int ncol,
                                    int K, int kc, int kb) {
    return *(const v16bf*)(Bt + (size_t)ncol * K + kc + kb);
}

// ---------------------------------------------------------------------------
// Fused GEMM: D = prelu(A1 @ B1 [+ A2 @ B2] + bias, alpha), D in bf16.
// Block = 128 threads (4 waves); wave w computes the 16x16 tile at
// (m0 = 16*blockIdx.x, n0 = 64*blockIdx.y + 16*w).  N (cols) fixed = 128.
// STAGE_B: async-copy the block's 64-column B window (16 KB/matrix) into LDS
// once, then serve all fragment reads from LDS (valid when K == HIDC).
// ---------------------------------------------------------------------------
template <bool A1_F32, bool DUAL, bool A2_F32, bool STAGE_B>
__global__ __launch_bounds__(128)
void gemm_wmma(const void* __restrict__ A1, const bf16_t* __restrict__ B1t,
               const void* __restrict__ A2, const bf16_t* __restrict__ B2t,
               const float* __restrict__ bias, const float* __restrict__ alpha,
               bf16_t* __restrict__ D, int M, int K, int lda1, int lda2) {
    // 16 KB per matrix when staging; 1 element when not (no LDS cost).
    __shared__ __align__(32) bf16_t sB[STAGE_B ? (DUAL ? 2 : 1) * 64 * HIDC : 1];

    const int lane = threadIdx.x & 31;
    const int wave = threadIdx.x >> 5;
    const int m0   = blockIdx.x * 16;
    const int nb0  = blockIdx.y * 64;          // block's column window
    const int n0   = nb0 + wave * 16;          // this wave's column tile
    const int row  = m0 + (lane & 15);
    const int sub  = (lane >> 4) * 16;  // A half-select within 32-wide K chunk
    const int col  = lane & 15;
    const int kb   = (lane >> 4) * 16;  // B half-select within 32-wide K chunk

    if constexpr (STAGE_B) {
        // 64 rows x 256 B = 1024 16-byte chunks per matrix; 8 per thread.
        const int t = threadIdx.x;
#pragma unroll
        for (int i = 0; i < 8; ++i) {
            const int chunk = t + i * 128;  // 0..1023
            const uint32_t l0 = (uint32_t)(uintptr_t)&sB[chunk * 8];
            const bf16_t* g0 = B1t + (size_t)nb0 * HIDC + chunk * 8;
            async_copy_b128_to_lds(l0, g0);
            if constexpr (DUAL) {
                const uint32_t l1 = (uint32_t)(uintptr_t)&sB[64 * HIDC + chunk * 8];
                const bf16_t* g1 = B2t + (size_t)nb0 * HIDC + chunk * 8;
                async_copy_b128_to_lds(l1, g1);
            }
        }
        wait_asynccnt0();
        __syncthreads();
    }

    const int bRowLocal = wave * 16 + col;     // row of Bt within LDS window

    v8f c = {};
#pragma unroll 4
    for (int kc = 0; kc < K; kc += 32) {
        v16bf a1 = load_a_frag<A1_F32>(A1, row, lda1, kc, sub);
        v16bf b1;
        if constexpr (STAGE_B)
            b1 = *(const v16bf*)&sB[bRowLocal * HIDC + kc + kb];
        else
            b1 = load_b_frag(B1t, n0 + col, K, kc, kb);
        c = __builtin_amdgcn_wmma_f32_16x16x32_bf16(
                false, a1, false, b1, (short)0, c, false, false);
        if constexpr (DUAL) {
            v16bf a2 = load_a_frag<A2_F32>(A2, row, lda2, kc, sub);
            v16bf b2;
            if constexpr (STAGE_B)
                b2 = *(const v16bf*)&sB[64 * HIDC + bRowLocal * HIDC + kc + kb];
            else
                b2 = load_b_frag(B2t, n0 + col, K, kc, kb);
            c = __builtin_amdgcn_wmma_f32_16x16x32_bf16(
                    false, a2, false, b2, (short)0, c, false, false);
        }
    }

    // Epilogue: bias + PReLU, store bf16. C/D layout: lane&15 = column,
    // VGPR r holds M = (lane>>4)*8 + r.
    const float bn = bias[n0 + col];
    const float an = alpha[n0 + col];
    const int mbase = m0 + (lane >> 4) * 8;
#pragma unroll
    for (int r = 0; r < 8; ++r) {
        float v = c[r] + bn;
        v = v > 0.0f ? v : an * v;
        D[(size_t)(mbase + r) * HIDC + (n0 + col)] = (bf16_t)v;
    }
}

// ---------------------------------------------------------------------------
// Weight convert + transpose: Wt[n*K + k] = bf16(W[k*N + n]).
// ---------------------------------------------------------------------------
__global__ void transpose_cvt(const float* __restrict__ W, bf16_t* __restrict__ Wt,
                              int K, int N) {
    int idx = blockIdx.x * blockDim.x + threadIdx.x;
    if (idx >= K * N) return;
    int k = idx / N, n = idx % N;
    Wt[(size_t)n * K + k] = (bf16_t)W[(size_t)k * N + n];
}

__global__ void zero_f32(float* __restrict__ p, int n) {
    int i = blockIdx.x * blockDim.x + threadIdx.x;
    if (i < n) p[i] = 0.0f;
}

// ---------------------------------------------------------------------------
// Edge scatter: agg[dst] += h[src] (bf16 gather, f32 atomic accumulate).
// One wave per edge; lane handles 4 channels (8 B gather -> 4 f32 atomics).
// global_prefetch on the next edge's source row hides gather latency.
// ---------------------------------------------------------------------------
__global__ __launch_bounds__(256)
void scatter_edges(const int* __restrict__ src, const int* __restrict__ dst,
                   const bf16_t* __restrict__ h, float* __restrict__ agg, int nE) {
    const int lane   = threadIdx.x & 31;
    const int gwave  = (blockIdx.x * blockDim.x + threadIdx.x) >> 5;
    const int nwaves = (gridDim.x * blockDim.x) >> 5;
    for (int e = gwave; e < nE; e += nwaves) {
        const int s = src[e];
        const int d = dst[e];
        const int en = e + nwaves;
        if (en < nE) {
            // gfx1250 global_prefetch_b8 of the next gather row
            __builtin_prefetch(h + (size_t)src[en] * HIDC + lane * 4, 0, 1);
        }
        const v4bf vv = *(const v4bf*)(h + (size_t)s * HIDC + lane * 4);
        float* ap = agg + (size_t)d * HIDC + lane * 4;
#pragma unroll
        for (int i = 0; i < 4; ++i) atomicAdd(ap + i, (float)vv[i]);
    }
}

// ---------------------------------------------------------------------------
// Global mean pool (sums + counts, mean folded into the head kernel).
// ---------------------------------------------------------------------------
__global__ void pool_sum(const bf16_t* __restrict__ h, const int* __restrict__ batch,
                         float* __restrict__ psum, int N) {
    int tid  = blockIdx.x * blockDim.x + threadIdx.x;
    int node = tid >> 5;
    int lane = tid & 31;
    if (node >= N) return;
    int g = batch[node];
    const v4bf vv = *(const v4bf*)(h + (size_t)node * HIDC + lane * 4);
    float* pp = psum + (size_t)g * HIDC + lane * 4;
#pragma unroll
    for (int i = 0; i < 4; ++i) atomicAdd(pp + i, (float)vv[i]);
}

__global__ void pool_count(const int* __restrict__ batch, float* __restrict__ cnt,
                           int N) {
    int i = blockIdx.x * blockDim.x + threadIdx.x;
    if (i < N) atomicAdd(&cnt[batch[i]], 1.0f);
}

// ---------------------------------------------------------------------------
// Final head: out[g] = [pooled_mean(g) ; fp_emb(g)] @ W_post + b_post.
// One 128-thread block (4 waves) per graph; shuffle + LDS reduction.
// ---------------------------------------------------------------------------
__global__ __launch_bounds__(128)
void final_head(const float* __restrict__ psum, const float* __restrict__ cnt,
                const bf16_t* __restrict__ fpe, const float* __restrict__ Wpost,
                const float* __restrict__ bpost, float* __restrict__ out) {
    const int g = blockIdx.x;
    const int c = threadIdx.x;           // 0..127
    float cn = cnt[g];
    cn = cn < 1.0f ? 1.0f : cn;
    float pooled = psum[(size_t)g * HIDC + c] / cn;
    float fe     = (float)fpe[(size_t)g * HIDC + c];
    float v = pooled * Wpost[c] + fe * Wpost[HIDC + c];
#pragma unroll
    for (int off = 16; off > 0; off >>= 1) v += __shfl_down(v, off, 32);
    __shared__ float sred[4];
    if ((threadIdx.x & 31) == 0) sred[threadIdx.x >> 5] = v;
    __syncthreads();
    if (threadIdx.x == 0)
        out[g] = sred[0] + sred[1] + sred[2] + sred[3] + bpost[0];
}

// ---------------------------------------------------------------------------
// Orchestration
// ---------------------------------------------------------------------------
extern "C" void kernel_launch(void* const* d_in, const int* in_sizes, int n_in,
                              void* d_out, int out_size, void* d_ws, size_t ws_size,
                              hipStream_t stream) {
    const float* x      = (const float*)d_in[0];
    const float* fp     = (const float*)d_in[1];
    const int*   edge   = (const int*)d_in[2];
    const int*   batch  = (const int*)d_in[3];
    const float* W_pre  = (const float*)d_in[4];
    const float* b_pre  = (const float*)d_in[5];
    const float* a_pre  = (const float*)d_in[6];
    const float* Wl1    = (const float*)d_in[7];
    const float* bl1    = (const float*)d_in[8];
    const float* Wr1    = (const float*)d_in[9];
    const float* a1     = (const float*)d_in[10];
    const float* Wl2    = (const float*)d_in[11];
    const float* bl2    = (const float*)d_in[12];
    const float* Wr2    = (const float*)d_in[13];
    const float* a2     = (const float*)d_in[14];
    const float* W_fp   = (const float*)d_in[15];
    const float* b_fp   = (const float*)d_in[16];
    const float* a_fp   = (const float*)d_in[17];
    const float* W_post = (const float*)d_in[18];
    const float* b_post = (const float*)d_in[19];

    const int N      = in_sizes[3];          // 100000 nodes (multiple of 16)
    const int nE     = in_sizes[2] / 2;      // 1.6M edges
    const int FPDIM  = 2048;
    const int G      = in_sizes[1] / FPDIM;  // 512 graphs
    const int* src = edge;
    const int* dst = edge + nE;

    // ---- carve workspace -------------------------------------------------
    uint8_t* w = (uint8_t*)d_ws;
    auto carve = [&](size_t bytes) -> void* {
        void* p = (void*)w;
        w += (bytes + 255) & ~(size_t)255;
        return p;
    };
    bf16_t* h0    = (bf16_t*)carve((size_t)N * HIDC * sizeof(bf16_t)); // h0, later h2
    bf16_t* h1    = (bf16_t*)carve((size_t)N * HIDC * sizeof(bf16_t));
    float*  agg   = (float*) carve((size_t)N * HIDC * sizeof(float));
    bf16_t* WtPre = (bf16_t*)carve((size_t)HIDC * HIDC * sizeof(bf16_t));
    bf16_t* WtL1  = (bf16_t*)carve((size_t)HIDC * HIDC * sizeof(bf16_t));
    bf16_t* WtR1  = (bf16_t*)carve((size_t)HIDC * HIDC * sizeof(bf16_t));
    bf16_t* WtL2  = (bf16_t*)carve((size_t)HIDC * HIDC * sizeof(bf16_t));
    bf16_t* WtR2  = (bf16_t*)carve((size_t)HIDC * HIDC * sizeof(bf16_t));
    bf16_t* WtFp  = (bf16_t*)carve((size_t)FPDIM * HIDC * sizeof(bf16_t));
    float*  psum  = (float*) carve((size_t)G * HIDC * sizeof(float));
    float*  cnt   = (float*) carve((size_t)G * sizeof(float));
    bf16_t* fpe   = (bf16_t*)carve((size_t)G * HIDC * sizeof(bf16_t));
    (void)ws_size; (void)n_in; (void)out_size;

    // ---- 1. weight convert + transpose ----------------------------------
    const int wElems = HIDC * HIDC;
    transpose_cvt<<<(wElems + 255) / 256, 256, 0, stream>>>(W_pre, WtPre, HIDC, HIDC);
    transpose_cvt<<<(wElems + 255) / 256, 256, 0, stream>>>(Wl1,   WtL1,  HIDC, HIDC);
    transpose_cvt<<<(wElems + 255) / 256, 256, 0, stream>>>(Wr1,   WtR1,  HIDC, HIDC);
    transpose_cvt<<<(wElems + 255) / 256, 256, 0, stream>>>(Wl2,   WtL2,  HIDC, HIDC);
    transpose_cvt<<<(wElems + 255) / 256, 256, 0, stream>>>(Wr2,   WtR2,  HIDC, HIDC);
    transpose_cvt<<<(FPDIM * HIDC + 255) / 256, 256, 0, stream>>>(W_fp, WtFp, FPDIM, HIDC);

    const dim3 gemmGridN(N / 16, 2);   // 4 waves/block cover 4 of the 8 col tiles
    const dim3 gemmGridG(G / 16, 2);

    // ---- 2. pre-MP: h0 = prelu(x @ W_pre + b_pre) ------------------------
    gemm_wmma<true, false, false, true><<<gemmGridN, 128, 0, stream>>>(
        x, WtPre, nullptr, nullptr, b_pre, a_pre, h0, N, HIDC, HIDC, 0);

    // ---- 3. conv1: agg = scatter(h0); h1 = prelu(agg@Wl1 + bl1 + h0@Wr1) -
    zero_f32<<<(N * HIDC + 255) / 256, 256, 0, stream>>>(agg, N * HIDC);
    scatter_edges<<<8192, 256, 0, stream>>>(src, dst, h0, agg, nE);
    gemm_wmma<true, true, false, true><<<gemmGridN, 128, 0, stream>>>(
        agg, WtL1, h0, WtR1, bl1, a1, h1, N, HIDC, HIDC, HIDC);

    // ---- 4. conv2: h2 (reusing h0 buffer) --------------------------------
    zero_f32<<<(N * HIDC + 255) / 256, 256, 0, stream>>>(agg, N * HIDC);
    scatter_edges<<<8192, 256, 0, stream>>>(src, dst, h1, agg, nE);
    gemm_wmma<true, true, false, true><<<gemmGridN, 128, 0, stream>>>(
        agg, WtL2, h1, WtR2, bl2, a2, h0, N, HIDC, HIDC, HIDC);

    // ---- 5. global mean pool --------------------------------------------
    zero_f32<<<(G * HIDC + 255) / 256, 256, 0, stream>>>(psum, G * HIDC);
    zero_f32<<<(G + 255) / 256, 256, 0, stream>>>(cnt, G);
    pool_sum<<<(N * 32 + 255) / 256, 256, 0, stream>>>(h0, batch, psum, N);
    pool_count<<<(N + 255) / 256, 256, 0, stream>>>(batch, cnt, N);

    // ---- 6. fingerprint head: fpe = prelu(fp @ W_fp + b_fp) --------------
    //         (K = 2048: B window too large to stage -> direct global path)
    gemm_wmma<true, false, false, false><<<gemmGridG, 128, 0, stream>>>(
        fp, WtFp, nullptr, nullptr, b_fp, a_fp, fpe, G, FPDIM, FPDIM, 0);

    // ---- 7. post-MP head -------------------------------------------------
    final_head<<<G, 128, 0, stream>>>(psum, cnt, fpe, W_post, b_post, (float*)d_out);
}
